// GNN_3934190043286
// MI455X (gfx1250) — matrix operand order repaired and checked by reading, not compiled, not run on hardware
//
#include <hip/hip_runtime.h>

// ---------------- constants from the reference ----------------
#define GH   4          // GAT heads
#define GC   32         // channels per head / hidden_dim
#define GHC  128        // GH*GC (also F_IN)
#define GNG  64         // graphs
#define GOUT 10         // classes

typedef __attribute__((ext_vector_type(16))) _Float16 v16h;
typedef __attribute__((ext_vector_type(8)))  _Float16 v8h;
typedef __attribute__((ext_vector_type(8)))  float    v8f;

// ---------------- WMMA GEMM: C[M x Nc] = A[M x K](f16) * Bt[Nc x K]^T (f16) ----------------
// A row-major [M][K]; Bt is B transposed, row-major [Nc][K].  K % 32 == 0, Nc % 16 == 0.
// One wave computes one 16x16 output tile, iterating K in steps of 32 via
// v_wmma_f32_16x16x32_f16 (f32 accumulate).

__device__ inline v16h load_a_tile(const _Float16* __restrict__ A, int M, int K,
                                   int row0, int k0, int lane) {
  // A tile, contiguous k order used here (same as B): lanes 0-15 row M=lane hold K k0..k0+15 in halves 0..15;
  // lanes 16-31 same rows, K k0+16..k0+31.
  int half = lane >> 4;
  int r = row0 + (lane & 15);
  if (r >= M) r = M - 1;                 // duplicate-load tail rows; store is guarded
  const _Float16* p = A + (size_t)r * K + k0 + half * 16;
  v8h lo = *(const v8h*)(p);
  v8h hi = *(const v8h*)(p + 8);
  return __builtin_shufflevector(lo, hi, 0,1,2,3,4,5,6,7,8,9,10,11,12,13,14,15);
}

__device__ inline v16h load_b_tile(const _Float16* __restrict__ Bt, int K,
                                   int col0, int k0, int lane) {
  // B tile, same contiguous k order as A: lanes 0-15 col N=lane hold K k0..k0+15 in halves 0..15;
  // lanes 16-31 hold K k0+16..k0+31.
  int half = lane >> 4;
  int c = col0 + (lane & 15);
  const _Float16* p = Bt + (size_t)c * K + k0 + half * 16;
  return *(const v16h*)(p);
}

__global__ __launch_bounds__(32)
void gemm_f16_wmma(const _Float16* __restrict__ A, const _Float16* __restrict__ Bt,
                   float* __restrict__ C, int M, int K, int Nc) {
  int lane = threadIdx.x;
  int row0 = blockIdx.x * 16;
  int col0 = blockIdx.y * 16;
  v8f acc = {};
  for (int k0 = 0; k0 < K; k0 += 32) {
    v16h a = load_a_tile(A, M, K, row0, k0, lane);
    v16h b = load_b_tile(Bt, K, col0, k0, lane);
    acc = __builtin_amdgcn_wmma_f32_16x16x32_f16(
        false, a, false, b, (short)0, acc, false, false);
  }
  // f32 C/D layout: VGPR r -> M = row0 + r + 8*(lane>=16), N = col0 + (lane&15)
  int half = lane >> 4;
  int nc = col0 + (lane & 15);
#pragma unroll
  for (int r = 0; r < 8; ++r) {
    int mr = row0 + half * 8 + r;
    if (mr < M) C[(size_t)mr * Nc + nc] = acc[r];
  }
}

// ---------------- small utility kernels ----------------
__global__ void k_fillf(float* p, float v, int n) {
  int t = blockIdx.x * blockDim.x + threadIdx.x;
  if (t < n) p[t] = v;
}
__global__ void k_f32_to_f16(const float* __restrict__ in, _Float16* __restrict__ out, int n) {
  int t = blockIdx.x * blockDim.x + threadIdx.x;
  if (t < n) out[t] = (_Float16)in[t];
}
// pack W [K x Nc] f32 row-major -> Wt [Nc x K] f16 row-major (transpose + cvt)
__global__ void k_pack_wt(const float* __restrict__ W, _Float16* __restrict__ Wt,
                          int K, int Nc) {
  int t = blockIdx.x * blockDim.x + threadIdx.x;
  if (t >= K * Nc) return;
  int k = t / Nc, n0 = t % Nc;
  Wt[(size_t)n0 * K + k] = (_Float16)W[t];
}

__device__ inline void atomicMaxF(float* addr, float val) {
  unsigned int* ua = (unsigned int*)addr;
  unsigned int old = *ua;
  for (;;) {
    float f = __uint_as_float(old);
    if (f >= val) break;
    unsigned int assumed = old;
    old = atomicCAS(ua, assumed, __float_as_uint(val));
    if (old == assumed) break;
  }
}

// ---------------- GAT pieces ----------------
// per-node per-head attention scores: a_s = sum_c h[n,h,c]*att_src[h,c]; same a_d
__global__ void k_att_scores(const float* __restrict__ h, const float* __restrict__ asw,
                             const float* __restrict__ adw, float* __restrict__ as_o,
                             float* __restrict__ ad_o, int n) {
  int t = blockIdx.x * blockDim.x + threadIdx.x;
  if (t >= n * GH) return;
  int node = t / GH, hh = t % GH;
  const float* hp = h + (size_t)node * GHC + hh * GC;
  float s = 0.f, d = 0.f;
#pragma unroll
  for (int c = 0; c < GC; ++c) {
    float v = hp[c];
    s += v * asw[hh * GC + c];
    d += v * adw[hh * GC + c];
  }
  as_o[t] = s;
  ad_o[t] = d;
}

__device__ inline void edge_sd(const int* __restrict__ src, const int* __restrict__ dst,
                               int E, int e, int& s, int& d) {
  if (e < E) { s = src[e]; d = dst[e]; }
  else       { s = e - E; d = e - E; }   // appended self-loops
}

__global__ void k_edge_max(const int* __restrict__ src, const int* __restrict__ dst,
                           const float* __restrict__ as_o, const float* __restrict__ ad_o,
                           float* __restrict__ m, int E, int Nn) {
  int t = blockIdx.x * blockDim.x + threadIdx.x;
  int EE = E + Nn;
  if (t >= EE * GH) return;
  int e = t / GH, hh = t % GH, s, d;
  edge_sd(src, dst, E, e, s, d);
  float v = as_o[s * GH + hh] + ad_o[d * GH + hh];
  v = v > 0.f ? v : 0.2f * v;                     // leaky_relu(0.2)
  atomicMaxF(&m[d * GH + hh], v);
}

__global__ void k_edge_expsum(const int* __restrict__ src, const int* __restrict__ dst,
                              const float* __restrict__ as_o, const float* __restrict__ ad_o,
                              const float* __restrict__ m, float* __restrict__ ebuf,
                              float* __restrict__ z, int E, int Nn) {
  int t = blockIdx.x * blockDim.x + threadIdx.x;
  int EE = E + Nn;
  if (t >= EE * GH) return;
  int e = t / GH, hh = t % GH, s, d;
  edge_sd(src, dst, E, e, s, d);
  float v = as_o[s * GH + hh] + ad_o[d * GH + hh];
  v = v > 0.f ? v : 0.2f * v;
  float ex = __expf(v - m[d * GH + hh]);
  ebuf[t] = ex;
  atomicAdd(&z[d * GH + hh], ex);
}

__global__ void k_edge_aggregate(const int* __restrict__ src, const int* __restrict__ dst,
                                 const float* __restrict__ ebuf, const float* __restrict__ z,
                                 const float* __restrict__ hfeat, float* __restrict__ out,
                                 int E, int Nn) {
  int t = blockIdx.x * blockDim.x + threadIdx.x;
  int EE = E + Nn;
  if (t >= EE * GH) return;
  int e = t / GH, hh = t % GH, s, d;
  edge_sd(src, dst, E, e, s, d);
  float alpha = ebuf[t] / (z[d * GH + hh] + 1e-16f);
  const float* hs = hfeat + (size_t)s * GHC + hh * GC;
  float* op = out + (size_t)d * GHC + hh * GC;
#pragma unroll
  for (int c = 0; c < GC; ++c) atomicAdd(&op[c], hs[c] * alpha);
}

// out = relu(acc + bias), write f32 + f16 copy
__global__ void k_bias_relu_cvt(const float* __restrict__ acc, const float* __restrict__ bias,
                                float* __restrict__ xo, _Float16* __restrict__ xh,
                                int n, int F) {
  int t = blockIdx.x * blockDim.x + threadIdx.x;
  if (t >= n * F) return;
  float v = acc[t] + bias[t % F];
  v = v > 0.f ? v : 0.f;
  xo[t] = v;
  xh[t] = (_Float16)v;
}

// out = relu(o1 + o2 + bias), write f32 + f16 copy
__global__ void k_add2_bias_relu_cvt(const float* __restrict__ o1, const float* __restrict__ o2,
                                     const float* __restrict__ bias, float* __restrict__ xo,
                                     _Float16* __restrict__ xh, int n, int F) {
  int t = blockIdx.x * blockDim.x + threadIdx.x;
  if (t >= n * F) return;
  float v = o1[t] + o2[t] + bias[t % F];
  v = v > 0.f ? v : 0.f;
  xo[t] = v;
  xh[t] = (_Float16)v;
}

// ---------------- GraphConv / SAGE pieces ----------------
__global__ void k_scatter_sum_feat(const int* __restrict__ src, const int* __restrict__ dst,
                                   const float* __restrict__ x, float* __restrict__ agg,
                                   int E, int F) {
  int t = blockIdx.x * blockDim.x + threadIdx.x;
  if (t >= E * F) return;
  int e = t / F, f = t % F;
  atomicAdd(&agg[(size_t)dst[e] * F + f], x[(size_t)src[e] * F + f]);
}
__global__ void k_deg(const int* __restrict__ dst, float* __restrict__ deg, int E) {
  int t = blockIdx.x * blockDim.x + threadIdx.x;
  if (t < E) atomicAdd(&deg[dst[t]], 1.0f);
}
__global__ void k_div_deg(float* __restrict__ mean, const float* __restrict__ deg,
                          int n, int F) {
  int t = blockIdx.x * blockDim.x + threadIdx.x;
  if (t >= n * F) return;
  mean[t] /= fmaxf(deg[t / F], 1.0f);
}

// ---------------- pooling + head ----------------
__global__ void k_pool_scatter(const int* __restrict__ batch, const float* __restrict__ x,
                               float* __restrict__ pool, float* __restrict__ cnt,
                               int n, int F) {
  int t = blockIdx.x * blockDim.x + threadIdx.x;
  if (t >= n * F) return;
  int node = t / F, f = t % F;
  int g = batch[node];
  atomicAdd(&pool[(size_t)g * F + f], x[t]);
  if (f == 0) atomicAdd(&cnt[g], 1.0f);
}

__global__ void k_mlp_head(const float* __restrict__ pool, const float* __restrict__ cnt,
                           const float* __restrict__ Wf1, const float* __restrict__ bf1,
                           const float* __restrict__ Wf2, const float* __restrict__ bf2,
                           float* __restrict__ out) {
  int g = blockIdx.x * blockDim.x + threadIdx.x;
  if (g >= GNG) return;
  float gi[GC], xr[GC];
  float inv = 1.0f / fmaxf(cnt[g], 1.0f);
#pragma unroll
  for (int c = 0; c < GC; ++c) gi[c] = pool[g * GC + c] * inv;
  for (int j = 0; j < GC; ++j) {
    float a = bf1[j];
#pragma unroll
    for (int c = 0; c < GC; ++c) a += gi[c] * Wf1[c * GC + j];
    xr[j] = a > 0.f ? a : 0.f;
  }
  for (int o = 0; o < GOUT; ++o) {
    float a = bf2[o];
#pragma unroll
    for (int j = 0; j < GC; ++j) a += xr[j] * Wf2[j * GOUT + o];
    out[g * GOUT + o] = a;
  }
}

// ---------------- host driver ----------------
static inline int blocks_for(long long n, int bs) { return (int)((n + bs - 1) / bs); }

extern "C" void kernel_launch(void* const* d_in, const int* in_sizes, int n_in,
                              void* d_out, int out_size, void* d_ws, size_t ws_size,
                              hipStream_t stream) {
  (void)n_in; (void)out_size; (void)ws_size;
  const float* x   = (const float*)d_in[0];
  const int*   ei  = (const int*)d_in[1];
  const int*   bat = (const int*)d_in[2];
  const float* W1  = (const float*)d_in[3];
  const float* as1 = (const float*)d_in[4];
  const float* ad1 = (const float*)d_in[5];
  const float* b1  = (const float*)d_in[6];
  const float* W2  = (const float*)d_in[7];
  const float* as2 = (const float*)d_in[8];
  const float* ad2 = (const float*)d_in[9];
  const float* b2  = (const float*)d_in[10];
  const float* W3r = (const float*)d_in[11];
  const float* W3l = (const float*)d_in[12];
  const float* b3  = (const float*)d_in[13];
  const float* W4l = (const float*)d_in[14];
  const float* b4l = (const float*)d_in[15];
  const float* W4r = (const float*)d_in[16];
  const float* Wf1 = (const float*)d_in[17];
  const float* bf1 = (const float*)d_in[18];
  const float* Wf2 = (const float*)d_in[19];
  const float* bf2 = (const float*)d_in[20];
  float* out = (float*)d_out;

  const int Nn = in_sizes[0] / GHC;  // 50000
  const int E  = in_sizes[1] / 2;    // 800000
  const int* src = ei;
  const int* dst = ei + E;

  // ---- workspace sub-allocation (256-B aligned) ----
  char* wp = (char*)d_ws;
  auto walloc = [&](size_t bytes) -> void* {
    void* p = (void*)wp;
    wp += (bytes + 255) & ~(size_t)255;
    return p;
  };
  float*    XF   = (float*)   walloc((size_t)Nn * GHC * 4);      // f32 features
  _Float16* XH   = (_Float16*)walloc((size_t)Nn * GHC * 2);      // f16 features
  _Float16* XH2  = (_Float16*)walloc((size_t)Nn * GHC * 2);      // f16 aggregate
  float*    HB   = (float*)   walloc((size_t)Nn * GHC * 4);      // gemm outputs
  float*    ACC  = (float*)   walloc((size_t)Nn * GHC * 4);      // scatter accum
  _Float16* WTA  = (_Float16*)walloc((size_t)GHC * GHC * 2);
  _Float16* WTB  = (_Float16*)walloc((size_t)GHC * GHC * 2);
  float*    ASB  = (float*)   walloc((size_t)Nn * GH * 4);
  float*    ADB  = (float*)   walloc((size_t)Nn * GH * 4);
  float*    MB   = (float*)   walloc((size_t)Nn * GH * 4);
  float*    ZB   = (float*)   walloc((size_t)Nn * GH * 4);
  float*    EBUF = (float*)   walloc((size_t)(E + Nn) * GH * 4);
  float*    DEG  = (float*)   walloc((size_t)Nn * 4);
  float*    POOL = (float*)   walloc((size_t)GNG * GC * 4);
  float*    CNT  = (float*)   walloc((size_t)GNG * 4);

  const int BS = 256;
  const int tm = (Nn + 15) / 16;
  const int EEH = (E + Nn) * GH;

  // GAT layer: input XH (f16, K=128) -> XF/XH (relu'ed output)
  auto run_gat = [&](const float* W, const float* asw, const float* adw, const float* bias) {
    k_pack_wt<<<blocks_for(GHC * GHC, BS), BS, 0, stream>>>(W, WTA, GHC, GHC);
    gemm_f16_wmma<<<dim3(tm, GHC / 16), 32, 0, stream>>>(XH, WTA, HB, Nn, GHC, GHC);
    k_att_scores<<<blocks_for((long long)Nn * GH, BS), BS, 0, stream>>>(HB, asw, adw, ASB, ADB, Nn);
    k_fillf<<<blocks_for((long long)Nn * GH, BS), BS, 0, stream>>>(MB, -1e30f, Nn * GH);
    k_fillf<<<blocks_for((long long)Nn * GH, BS), BS, 0, stream>>>(ZB, 0.f, Nn * GH);
    k_fillf<<<blocks_for((long long)Nn * GHC, BS), BS, 0, stream>>>(ACC, 0.f, Nn * GHC);
    k_edge_max<<<blocks_for(EEH, BS), BS, 0, stream>>>(src, dst, ASB, ADB, MB, E, Nn);
    k_edge_expsum<<<blocks_for(EEH, BS), BS, 0, stream>>>(src, dst, ASB, ADB, MB, EBUF, ZB, E, Nn);
    k_edge_aggregate<<<blocks_for(EEH, BS), BS, 0, stream>>>(src, dst, EBUF, ZB, HB, ACC, E, Nn);
    k_bias_relu_cvt<<<blocks_for((long long)Nn * GHC, BS), BS, 0, stream>>>(ACC, bias, XF, XH, Nn, GHC);
  };

  // ---- stage 0: x -> f16 ----
  k_f32_to_f16<<<blocks_for((long long)Nn * GHC, BS), BS, 0, stream>>>(x, XH, Nn * GHC);

  // ---- GAT1, GAT2 ----
  run_gat(W1, as1, ad1, b1);
  run_gat(W2, as2, ad2, b2);

  // ---- GraphConv: relu(segsum(x[src]) @ W3r + b3 + x @ W3l) ----
  k_fillf<<<blocks_for((long long)Nn * GHC, BS), BS, 0, stream>>>(ACC, 0.f, Nn * GHC);
  k_scatter_sum_feat<<<blocks_for((long long)E * GHC, BS), BS, 0, stream>>>(src, dst, XF, ACC, E, GHC);
  k_f32_to_f16<<<blocks_for((long long)Nn * GHC, BS), BS, 0, stream>>>(ACC, XH2, Nn * GHC);
  k_pack_wt<<<blocks_for(GHC * GC, BS), BS, 0, stream>>>(W3r, WTA, GHC, GC);
  k_pack_wt<<<blocks_for(GHC * GC, BS), BS, 0, stream>>>(W3l, WTB, GHC, GC);
  gemm_f16_wmma<<<dim3(tm, GC / 16), 32, 0, stream>>>(XH2, WTA, HB,           Nn, GHC, GC);
  gemm_f16_wmma<<<dim3(tm, GC / 16), 32, 0, stream>>>(XH,  WTB, HB + (size_t)Nn * GC, Nn, GHC, GC);
  k_add2_bias_relu_cvt<<<blocks_for((long long)Nn * GC, BS), BS, 0, stream>>>(
      HB, HB + (size_t)Nn * GC, b3, XF, XH, Nn, GC);

  // ---- SAGEConv: relu(mean(x[src]) @ W4l + b4l + x @ W4r) ----
  k_fillf<<<blocks_for(Nn, BS), BS, 0, stream>>>(DEG, 0.f, Nn);
  k_deg<<<blocks_for(E, BS), BS, 0, stream>>>(dst, DEG, E);
  k_fillf<<<blocks_for((long long)Nn * GC, BS), BS, 0, stream>>>(ACC, 0.f, Nn * GC);
  k_scatter_sum_feat<<<blocks_for((long long)E * GC, BS), BS, 0, stream>>>(src, dst, XF, ACC, E, GC);
  k_div_deg<<<blocks_for((long long)Nn * GC, BS), BS, 0, stream>>>(ACC, DEG, Nn, GC);
  k_f32_to_f16<<<blocks_for((long long)Nn * GC, BS), BS, 0, stream>>>(ACC, XH2, Nn * GC);
  k_pack_wt<<<blocks_for(GC * GC, BS), BS, 0, stream>>>(W4l, WTA, GC, GC);
  k_pack_wt<<<blocks_for(GC * GC, BS), BS, 0, stream>>>(W4r, WTB, GC, GC);
  gemm_f16_wmma<<<dim3(tm, GC / 16), 32, 0, stream>>>(XH2, WTA, HB,           Nn, GC, GC);
  gemm_f16_wmma<<<dim3(tm, GC / 16), 32, 0, stream>>>(XH,  WTB, HB + (size_t)Nn * GC, Nn, GC, GC);
  k_add2_bias_relu_cvt<<<blocks_for((long long)Nn * GC, BS), BS, 0, stream>>>(
      HB, HB + (size_t)Nn * GC, b4l, XF, XH, Nn, GC);

  // ---- global mean pool + MLP head ----
  k_fillf<<<blocks_for(GNG * GC, BS), BS, 0, stream>>>(POOL, 0.f, GNG * GC);
  k_fillf<<<blocks_for(GNG, BS), BS, 0, stream>>>(CNT, 0.f, GNG);
  k_pool_scatter<<<blocks_for((long long)Nn * GC, BS), BS, 0, stream>>>(bat, XF, POOL, CNT, Nn, GC);
  k_mlp_head<<<2, 32, 0, stream>>>(POOL, CNT, Wf1, bf1, Wf2, bf2, out);
}
